// MiniEncoder_PN2Seg_41532333752451
// MI455X (gfx1250) — matrix operand order repaired and checked by reading, not compiled, not run on hardware
//
#include <hip/hip_runtime.h>
#include <hip/hip_bf16.h>
#include <cstdint>
#include <cstddef>

typedef __bf16 bf16;
typedef __attribute__((ext_vector_type(16))) __bf16 v16bf;
typedef __attribute__((ext_vector_type(8)))  __bf16 v8bf;
typedef __attribute__((ext_vector_type(8)))  float   v8f;

// ---------------------------------------------------------------------------
// Elementwise f32 -> bf16
__global__ void k_tobf(const float* __restrict__ x, bf16* __restrict__ y, long long n) {
  long long t = (long long)blockIdx.x * blockDim.x + threadIdx.x;
  if (t < n) y[t] = (bf16)x[t];
}

// ---------------------------------------------------------------------------
// Farthest point sampling: one block per batch; dist kept in LDS.
__global__ void k_fps(const float* __restrict__ xyz, int N, int npoint, int* __restrict__ outIdx) {
  int b = blockIdx.x, tid = threadIdx.x;
  __shared__ float dist[4096];
  __shared__ float rv[256];
  __shared__ int   ri[256];
  const float* p = xyz + (size_t)b * N * 3;
  for (int n = tid; n < N; n += 256) dist[n] = 1e10f;
  __syncthreads();
  int far = 0;
  for (int it = 0; it < npoint; ++it) {
    if (tid == 0) outIdx[b * npoint + it] = far;
    float cx = p[far * 3 + 0], cy = p[far * 3 + 1], cz = p[far * 3 + 2];
    float bm = -1.0f; int bi = 0;
    for (int n = tid; n < N; n += 256) {
      float dx = p[n * 3 + 0] - cx, dy = p[n * 3 + 1] - cy, dz = p[n * 3 + 2] - cz;
      float d  = dx * dx + dy * dy + dz * dz;
      float dm = fminf(dist[n], d);
      dist[n] = dm;
      if (dm > bm) { bm = dm; bi = n; }
    }
    rv[tid] = bm; ri[tid] = bi;
    __syncthreads();
    for (int s = 128; s > 0; s >>= 1) {
      if (tid < s) {
        if (rv[tid + s] > rv[tid] || (rv[tid + s] == rv[tid] && ri[tid + s] < ri[tid])) {
          rv[tid] = rv[tid + s]; ri[tid] = ri[tid + s];
        }
      }
      __syncthreads();
    }
    far = ri[0];
    __syncthreads();
  }
}

// Gather xyz rows by index: out[b,s,:] = xyz[b, idx[b,s], :]
__global__ void k_gather3(const float* __restrict__ xyz, const int* __restrict__ idx,
                          float* __restrict__ out, int S, int N, int B) {
  int t = blockIdx.x * blockDim.x + threadIdx.x;
  if (t >= B * S * 3) return;
  int c = t % 3, bs = t / 3, b = bs / S;
  out[t] = xyz[((size_t)b * N + idx[bs]) * 3 + c];
}

// Ball query: first-K (index order) neighbors within radius, pad with first.
__global__ void k_ballq(const float* __restrict__ xyz, const float* __restrict__ cen,
                        int N, int S, int K, float r2, int B, int* __restrict__ gidx) {
  int t = blockIdx.x * blockDim.x + threadIdx.x;
  if (t >= B * S) return;
  int b = t / S;
  const float* p = xyz + (size_t)b * N * 3;
  float cx = cen[t * 3 + 0], cy = cen[t * 3 + 1], cz = cen[t * 3 + 2];
  int* out = gidx + (size_t)t * K;
  int cnt = 0, first = -1;
  for (int n = 0; n < N && cnt < K; ++n) {
    float dx = p[n * 3 + 0] - cx, dy = p[n * 3 + 1] - cy, dz = p[n * 3 + 2] - cz;
    if (dx * dx + dy * dy + dz * dz <= r2) {
      if (first < 0) first = n;
      out[cnt++] = n;
    }
  }
  if (first < 0) first = 0;
  for (; cnt < K; ++cnt) out[cnt] = first;
}

// Build grouped features: [feats[idx] (Cf) | xyz[idx]-center (3) | zero pad]
__global__ void k_group(const bf16* __restrict__ feats, int Cf, int ldF,
                        const float* __restrict__ xyz, const float* __restrict__ cen,
                        const int* __restrict__ gidx, bf16* __restrict__ out,
                        int Kpad, int N, int S, int K, int B) {
  long long t = (long long)blockIdx.x * blockDim.x + threadIdx.x;
  long long total = (long long)B * S * K;
  if (t >= total) return;
  long long bs = t / K;
  int b = (int)(bs / S);
  int idx = gidx[t];
  bf16* o = out + t * Kpad;
  const bf16* f = feats + ((size_t)b * N + idx) * ldF;
  for (int c = 0; c < Cf; ++c) o[c] = f[c];
  float cx = cen[bs * 3 + 0], cy = cen[bs * 3 + 1], cz = cen[bs * 3 + 2];
  const float* q = xyz + ((size_t)b * N + idx) * 3;
  o[Cf + 0] = (bf16)(q[0] - cx);
  o[Cf + 1] = (bf16)(q[1] - cy);
  o[Cf + 2] = (bf16)(q[2] - cz);
  for (int c = Cf + 3; c < Kpad; ++c) o[c] = (bf16)0.0f;
}

// Pack weight (Cout,Cin) f32 -> padded (Npad,Kpad) bf16
__global__ void k_packw(const float* __restrict__ W, int Nw, int Kw,
                        bf16* __restrict__ o, int Npad, int Kpad) {
  long long t = (long long)blockIdx.x * blockDim.x + threadIdx.x;
  if (t >= (long long)Npad * Kpad) return;
  int n = (int)(t / Kpad), k = (int)(t % Kpad);
  o[t] = (n < Nw && k < Kw) ? (bf16)W[(size_t)n * Kw + k] : (bf16)0.0f;
}

// ---------------------------------------------------------------------------
// Core WMMA GEMM: out[M,ldOut] = A[M,Kpad] * W[Npad,Kpad]^T + bias
// Each wave computes a 32 x (JN*16) block: 2 M-tiles x JN N-tiles, so every
// B fragment feeds 2 WMMAs and every A fragment feeds JN WMMAs (8 WMMAs per
// K-step at JN=4). Single fragment slot (no rotation copies, no spills);
// wave id is readfirstlane'd so all loop control is scalar and EXEC stays
// all-ones around the WMMAs. Register budget ~130 VGPRs.
static __device__ __forceinline__ v16bf mergeA(v8bf a0, v8bf a1) {
  v16bf a;
#pragma unroll
  for (int i = 0; i < 8; ++i) { a[i] = a0[i]; a[i + 8] = a1[i]; }
  return a;
}

template <int JN>
__global__ void k_gemmT(const bf16* __restrict__ A, const bf16* __restrict__ W,
                        const float* __restrict__ bias, float* __restrict__ out,
                        int M, int Kpad, int colBase, int ngroups, int Nreal, int ldOut) {
  int gw = (blockIdx.x * blockDim.x + threadIdx.x) >> 5;
  gw = __builtin_amdgcn_readfirstlane(gw);          // wave-uniform -> SGPR
  int mt2 = M >> 5;                                  // 32-row strips
  if (gw >= mt2 * ngroups) return;
  int tm = gw % mt2;                // consecutive waves -> consecutive M strips
  int tg = gw / mt2;                // N-group shared within a block
  int lane = threadIdx.x & 31;
  int lo = lane & 15, hi = lane >> 4;

  const bf16* Ar0 = A + (size_t)(tm * 32 + lo) * Kpad;
  const bf16* Ar1 = Ar0 + (size_t)16 * Kpad;
  const bf16* W0  = W + (size_t)(colBase + tg * JN * 16 + lo) * Kpad;
  const size_t wstep = (size_t)16 * Kpad;

  v8f acc0[JN], acc1[JN];
#pragma unroll
  for (int j = 0; j < JN; ++j) {
    acc0[j] = (v8f){0.f,0.f,0.f,0.f,0.f,0.f,0.f,0.f};
    acc1[j] = (v8f){0.f,0.f,0.f,0.f,0.f,0.f,0.f,0.f};
  }

  for (int k = 0; k < Kpad; k += 32) {
    if (k + 96 < Kpad) {                            // scalar guard
      __builtin_prefetch(Ar0 + k + 96, 0, 0);
      __builtin_prefetch(Ar1 + k + 96, 0, 0);
    }
    // A 16x32 fragment: lanes 0-15 -> K {0..7,16..23}, lanes 16-31 -> K {8..15,24..31}
    v16bf a0 = mergeA(*(const v8bf*)(Ar0 + k + hi * 8),
                      *(const v8bf*)(Ar0 + k + 16 + hi * 8));
    v16bf a1 = mergeA(*(const v8bf*)(Ar1 + k + hi * 8),
                      *(const v8bf*)(Ar1 + k + 16 + hi * 8));
    // B 32x16 fragment: same K order as A (lanes 0-15 -> K {0..7,16..23}, lanes 16-31 -> K {8..15,24..31})
    v16bf b[JN];
#pragma unroll
    for (int j = 0; j < JN; ++j) b[j] = mergeA(*(const v8bf*)(W0 + j * wstep + k + hi * 8),
                                               *(const v8bf*)(W0 + j * wstep + k + 16 + hi * 8));
#pragma unroll
    for (int j = 0; j < JN; ++j) {
      acc0[j] = __builtin_amdgcn_wmma_f32_16x16x32_bf16(false, a0, false, b[j], (short)0, acc0[j], false, false);
      acc1[j] = __builtin_amdgcn_wmma_f32_16x16x32_bf16(false, a1, false, b[j], (short)0, acc1[j], false, false);
    }
  }

  int row0 = tm * 32 + hi * 8;      // C/D layout: VGPR r -> M = r (+8 for hi half)
#pragma unroll
  for (int j = 0; j < JN; ++j) {
    int col = colBase + tg * JN * 16 + j * 16 + lo;
    if (col < Nreal) {
      float bv = bias ? bias[col] : 0.0f;
#pragma unroll
      for (int r = 0; r < 8; ++r) {
        out[(size_t)(row0 + r) * ldOut + col]      = acc0[j][r] + bv;
        out[(size_t)(row0 + 16 + r) * ldOut + col] = acc1[j][r] + bv;
      }
    }
  }
}

// ---------------------------------------------------------------------------
// Per-channel mean / rsqrt(var+eps): one block per channel.
__global__ void k_bnstats(const float* __restrict__ x, long long M, int ld,
                          float* __restrict__ mean, float* __restrict__ rstd) {
  int c = blockIdx.x, tid = threadIdx.x;
  __shared__ float s1[256], s2[256];
  float a = 0.f, q = 0.f;
  for (long long m = tid; m < M; m += 256) {
    float v = x[m * ld + c];
    a += v; q += v * v;
  }
  s1[tid] = a; s2[tid] = q;
  __syncthreads();
  for (int s = 128; s > 0; s >>= 1) {
    if (tid < s) { s1[tid] += s1[tid + s]; s2[tid] += s2[tid + s]; }
    __syncthreads();
  }
  if (tid == 0) {
    float mu  = s1[0] / (float)M;
    float var = s2[0] / (float)M - mu * mu;
    mean[c] = mu;
    rstd[c] = rsqrtf(var + 1e-5f);
  }
}

// BN + ReLU + cast to bf16, zeroing the padded channels.
__global__ void k_bnrelu(const float* __restrict__ x, long long M, int ld, int C,
                         const float* __restrict__ mean, const float* __restrict__ rstd,
                         const float* __restrict__ g, const float* __restrict__ be,
                         bf16* __restrict__ y) {
  long long t = (long long)blockIdx.x * blockDim.x + threadIdx.x;
  if (t >= M * ld) return;
  int c = (int)(t % ld);
  float v = 0.0f;
  if (c < C) {
    v = (x[t] - mean[c]) * rstd[c] * g[c] + be[c];
    v = fmaxf(v, 0.0f);
  }
  y[t] = (bf16)v;
}

// Max over K grouped rows, write into concat buffer at column offset.
__global__ void k_maxpool(const bf16* __restrict__ in, int ldIn, int K, int C,
                          bf16* __restrict__ out, int ldOut, int colOff, int rows) {
  long long t = (long long)blockIdx.x * blockDim.x + threadIdx.x;
  if (t >= (long long)rows * C) return;
  int r = (int)(t / C), c = (int)(t % C);
  float m = -1e30f;
  const bf16* p = in + (size_t)r * K * ldIn + c;
  for (int k = 0; k < K; ++k) m = fmaxf(m, (float)p[(size_t)k * ldIn]);
  out[(size_t)r * ldOut + colOff + c] = (bf16)m;
}

// sa3 grouped input: [l2_xyz(3) | l2_points(512) | pad->544]
__global__ void k_sa3cat(const float* __restrict__ l2xyz, const bf16* __restrict__ l2p,
                         bf16* __restrict__ out) {
  long long t = (long long)blockIdx.x * blockDim.x + threadIdx.x;
  if (t >= (long long)512 * 544) return;
  int r = (int)(t / 544), c = (int)(t % 544);
  bf16 v;
  if (c < 3)        v = (bf16)l2xyz[r * 3 + c];
  else if (c < 515) v = l2p[(size_t)r * 512 + (c - 3)];
  else              v = (bf16)0.0f;
  out[t] = v;
}

// fp3 input: [l2_points(512) | broadcast l3_points(1024)] -> 1536
__global__ void k_fp3cat(const bf16* __restrict__ l2p, const bf16* __restrict__ l3,
                         bf16* __restrict__ out) {
  long long t = (long long)blockIdx.x * blockDim.x + threadIdx.x;
  if (t >= (long long)512 * 1536) return;
  int r = (int)(t / 1536), c = (int)(t % 1536), b = r / 128;
  out[t] = (c < 512) ? l2p[(size_t)r * 512 + c] : l3[(size_t)b * 1024 + (c - 512)];
}

// l0_in = [cls zeros(16) | xyz(3) | xyz(3)] -> 22 channels
__global__ void k_l0in(const float* __restrict__ xyz, bf16* __restrict__ out) {
  long long t = (long long)blockIdx.x * blockDim.x + threadIdx.x;
  if (t >= (long long)16384 * 22) return;
  int r = (int)(t / 22), c = (int)(t % 22);
  out[t] = (c < 16) ? (bf16)0.0f : (bf16)xyz[(size_t)r * 3 + ((c - 16) % 3)];
}

// 3-NN inverse-distance interpolation + concat: [p1(C1) | interp(C2) | pad]
__global__ void k_interp_concat(const float* __restrict__ xyz1, const float* __restrict__ xyz2,
                                const bf16* __restrict__ p1, int C1, int ld1,
                                const bf16* __restrict__ f2, int C2, int ld2,
                                bf16* __restrict__ out, int ldOut, int Nn, int S2, int B) {
  int t = blockIdx.x * blockDim.x + threadIdx.x;
  if (t >= B * Nn) return;
  int b = t / Nn;
  float px = xyz1[(size_t)t * 3 + 0], py = xyz1[(size_t)t * 3 + 1], pz = xyz1[(size_t)t * 3 + 2];
  const float* q = xyz2 + (size_t)b * S2 * 3;
  float d0 = 1e30f, d1 = 1e30f, d2 = 1e30f;
  int   i0 = 0,     i1 = 0,     i2 = 0;
  for (int s = 0; s < S2; ++s) {
    float dx = q[s * 3 + 0] - px, dy = q[s * 3 + 1] - py, dz = q[s * 3 + 2] - pz;
    float d = dx * dx + dy * dy + dz * dz;
    if (d < d0)      { d2 = d1; i2 = i1; d1 = d0; i1 = i0; d0 = d; i0 = s; }
    else if (d < d1) { d2 = d1; i2 = i1; d1 = d;  i1 = s; }
    else if (d < d2) { d2 = d;  i2 = s; }
  }
  float r0 = 1.0f / (d0 + 1e-8f), r1 = 1.0f / (d1 + 1e-8f), r2 = 1.0f / (d2 + 1e-8f);
  float rs = r0 + r1 + r2;
  r0 /= rs; r1 /= rs; r2 /= rs;
  bf16* o = out + (size_t)t * ldOut;
  const bf16* a = p1 + (size_t)t * ld1;
  for (int c = 0; c < C1; ++c) o[c] = a[c];
  const bf16* g0 = f2 + ((size_t)b * S2 + i0) * ld2;
  const bf16* g1 = f2 + ((size_t)b * S2 + i1) * ld2;
  const bf16* g2 = f2 + ((size_t)b * S2 + i2) * ld2;
  for (int c = 0; c < C2; ++c)
    o[C1 + c] = (bf16)(r0 * (float)g0[c] + r1 * (float)g1[c] + r2 * (float)g2[c]);
  for (int c = C1 + C2; c < ldOut; ++c) o[c] = (bf16)0.0f;
}

// ---------------------------------------------------------------------------
struct Layer { const float *W, *b, *g, *be; int cout, cin; };

static inline unsigned nblk(long long total, int bs) { return (unsigned)((total + bs - 1) / bs); }
static inline int rnd32(int x) { return (x + 31) & ~31; }

extern "C" void kernel_launch(void* const* d_in, const int* in_sizes, int n_in,
                              void* d_out, int out_size, void* d_ws, size_t ws_size,
                              hipStream_t stream) {
  (void)in_sizes; (void)n_in; (void)out_size; (void)ws_size;
  const int B = 4, N = 4096;

  // ---- parse inputs (setup_inputs dict order: xyz, then params insertion order;
  //      each layer dict in order W,b,g,be) ----
  int p = 0;
  const float* xyz = (const float*)d_in[p++];
  auto getL = [&](int cout, int cin) {
    Layer L{(const float*)d_in[p], (const float*)d_in[p + 1],
            (const float*)d_in[p + 2], (const float*)d_in[p + 3], cout, cin};
    p += 4;
    return L;
  };
  Layer sa1L[3][3];
  { int co[3][3] = {{32,32,64},{64,64,128},{64,96,128}};
    for (int br = 0; br < 3; ++br) { int ci = 6;
      for (int l = 0; l < 3; ++l) { sa1L[br][l] = getL(co[br][l], ci); ci = co[br][l]; } } }
  Layer sa2L[2][3];
  { int co[2][3] = {{128,128,256},{128,196,256}};
    for (int br = 0; br < 2; ++br) { int ci = 323;
      for (int l = 0; l < 3; ++l) { sa2L[br][l] = getL(co[br][l], ci); ci = co[br][l]; } } }
  Layer sa3L[3];
  { int co[3] = {256,512,1024}; int ci = 515;
    for (int l = 0; l < 3; ++l) { sa3L[l] = getL(co[l], ci); ci = co[l]; } }
  Layer fp3L[2]; { int co[2]={256,256}; int ci=1536; for(int l=0;l<2;++l){fp3L[l]=getL(co[l],ci); ci=co[l];} }
  Layer fp2L[2]; { int co[2]={256,128}; int ci=576;  for(int l=0;l<2;++l){fp2L[l]=getL(co[l],ci); ci=co[l];} }
  Layer fp1L[2]; { int co[2]={128,128}; int ci=150;  for(int l=0;l<2;++l){fp1L[l]=getL(co[l],ci); ci=co[l];} }
  Layer conv1L[1] = { getL(64, 128) };
  const float* c2W = (const float*)d_in[p++];
  const float* c2b = (const float*)d_in[p++];

  // ---- deterministic bump allocator over workspace ----
  uintptr_t base = (uintptr_t)d_ws;
  size_t off = 0;
  auto alloc = [&](size_t bytes) -> void* {
    void* r = (void*)(base + off);
    off = (off + bytes + 255) & ~(size_t)255;
    return r;
  };

  bf16*  xyz_bf = (bf16*) alloc((size_t)B * N * 3 * sizeof(bf16));
  int*   fidx1  = (int*)  alloc((size_t)B * 512 * sizeof(int));
  float* l1xyz  = (float*)alloc((size_t)B * 512 * 3 * sizeof(float));
  int*   fidx2  = (int*)  alloc((size_t)B * 128 * sizeof(int));
  float* l2xyz  = (float*)alloc((size_t)B * 128 * 3 * sizeof(float));
  int*   gidx   = (int*)  alloc((size_t)B * 512 * 128 * sizeof(int));
  bf16*  l1pts  = (bf16*) alloc((size_t)B * 512 * 320 * sizeof(bf16));
  bf16*  l2pts  = (bf16*) alloc((size_t)B * 128 * 512 * sizeof(bf16));
  bf16*  l3pts  = (bf16*) alloc((size_t)B * 1024 * sizeof(bf16));
  bf16*  l2p2   = (bf16*) alloc((size_t)B * 128 * 256 * sizeof(bf16));
  bf16*  l1p2   = (bf16*) alloc((size_t)B * 512 * 128 * sizeof(bf16));
  bf16*  l0in   = (bf16*) alloc((size_t)B * N * 22 * sizeof(bf16));
  float* meanb  = (float*)alloc(1024 * sizeof(float));
  float* rstdb  = (float*)alloc(1024 * sizeof(float));

  const long long MAXE = (long long)B * 512 * 128 * 128; // 33.55M elements
  bf16*  ping = (bf16*) alloc((size_t)MAXE * sizeof(bf16));
  bf16*  pong = (bf16*) alloc((size_t)MAXE * sizeof(bf16));
  float* fbuf = (float*)alloc((size_t)MAXE * sizeof(float));

  // ---- GEMM launcher: full groups of 4 N-tiles + (0|2)-tile remainder ----
  auto launch_gemm = [&](const bf16* A, const bf16* Wp, const float* bias, float* outp,
                         long long M, int kp, int cp, int Nreal, int ldOut) {
    int nt = cp / 16;
    int full = nt / 4, rem = nt % 4;
    long long mt2 = M / 32;         // every M here is a multiple of 32
    if (full)
      k_gemmT<4><<<nblk(mt2 * full * 32, 256), 256, 0, stream>>>(
          A, Wp, bias, outp, (int)M, kp, 0, full, Nreal, ldOut);
    if (rem == 2)
      k_gemmT<2><<<nblk(mt2 * 32, 256), 256, 0, stream>>>(
          A, Wp, bias, outp, (int)M, kp, full * 64, 1, Nreal, ldOut);
    else if (rem)   // generic fallback (unused for this network's shapes)
      for (int j = 0; j < rem; ++j)
        k_gemmT<1><<<nblk(mt2 * 32, 256), 256, 0, stream>>>(
            A, Wp, bias, outp, (int)M, kp, full * 64 + j * 16, 1, Nreal, ldOut);
  };

  // ---- MLP runner: GEMM(wmma) -> BN stats -> BN+ReLU(bf16); input in ping/pong ----
  auto run_mlp = [&](bf16* cur, long long M, int kpad0, const Layer* Ls, int nl) -> bf16* {
    int kp = kpad0;
    for (int l = 0; l < nl; ++l) {
      const Layer& L = Ls[l];
      int cp = rnd32(L.cout);
      bf16* wp = (bf16*)alloc((size_t)cp * kp * sizeof(bf16));
      k_packw<<<nblk((long long)cp * kp, 256), 256, 0, stream>>>(L.W, L.cout, L.cin, wp, cp, kp);
      launch_gemm(cur, wp, L.b, fbuf, M, kp, cp, L.cout, cp);
      k_bnstats<<<dim3(L.cout), 256, 0, stream>>>(fbuf, M, cp, meanb, rstdb);
      bf16* nxt = (cur == ping) ? pong : ping;
      k_bnrelu<<<nblk(M * cp, 256), 256, 0, stream>>>(fbuf, M, cp, L.cout, meanb, rstdb, L.g, L.be, nxt);
      cur = nxt;
      kp = cp;
    }
    return cur;
  };

  // =================== SA1 ===================
  k_fps<<<dim3(B), 256, 0, stream>>>(xyz, N, 512, fidx1);
  k_gather3<<<nblk(B * 512 * 3, 256), 256, 0, stream>>>(xyz, fidx1, l1xyz, 512, N, B);
  k_tobf<<<nblk((long long)B * N * 3, 256), 256, 0, stream>>>(xyz, xyz_bf, (long long)B * N * 3);

  { const float rad[3] = {0.1f, 0.2f, 0.4f};
    const int   ks[3]  = {32, 64, 128};
    const int   coff[3]= {0, 64, 192};
    for (int br = 0; br < 3; ++br) {
      int K = ks[br];
      k_ballq<<<nblk(B * 512, 128), 128, 0, stream>>>(xyz, l1xyz, N, 512, K, rad[br] * rad[br], B, gidx);
      long long M = (long long)B * 512 * K;
      k_group<<<nblk(M, 256), 256, 0, stream>>>(xyz_bf, 3, 3, xyz, l1xyz, gidx, ping, 32, N, 512, K, B);
      bf16* a = run_mlp(ping, M, 32, sa1L[br], 3);
      int cOut = sa1L[br][2].cout, cp = rnd32(cOut);
      k_maxpool<<<nblk((long long)B * 512 * cOut, 256), 256, 0, stream>>>(
          a, cp, K, cOut, l1pts, 320, coff[br], B * 512);
    } }

  // =================== SA2 ===================
  k_fps<<<dim3(B), 256, 0, stream>>>(l1xyz, 512, 128, fidx2);
  k_gather3<<<nblk(B * 128 * 3, 256), 256, 0, stream>>>(l1xyz, fidx2, l2xyz, 128, 512, B);

  { const float rad[2] = {0.4f, 0.8f};
    const int   ks[2]  = {64, 128};
    const int   coff[2]= {0, 256};
    for (int br = 0; br < 2; ++br) {
      int K = ks[br];
      k_ballq<<<nblk(B * 128, 128), 128, 0, stream>>>(l1xyz, l2xyz, 512, 128, K, rad[br] * rad[br], B, gidx);
      long long M = (long long)B * 128 * K;
      k_group<<<nblk(M, 256), 256, 0, stream>>>(l1pts, 320, 320, l1xyz, l2xyz, gidx, ping, 352, 512, 128, K, B);
      bf16* a = run_mlp(ping, M, 352, sa2L[br], 3);
      int cOut = sa2L[br][2].cout, cp = rnd32(cOut);
      k_maxpool<<<nblk((long long)B * 128 * cOut, 256), 256, 0, stream>>>(
          a, cp, K, cOut, l2pts, 512, coff[br], B * 128);
    } }

  // =================== SA3 ===================
  k_sa3cat<<<nblk((long long)512 * 544, 256), 256, 0, stream>>>(l2xyz, l2pts, ping);
  { bf16* a = run_mlp(ping, 512, 544, sa3L, 3);
    k_maxpool<<<nblk((long long)B * 1024, 256), 256, 0, stream>>>(a, 1024, 128, 1024, l3pts, 1024, 0, B); }

  // =================== FP3 ===================
  k_fp3cat<<<nblk((long long)512 * 1536, 256), 256, 0, stream>>>(l2pts, l3pts, ping);
  { bf16* a = run_mlp(ping, 512, 1536, fp3L, 2);
    hipMemcpyAsync(l2p2, a, (size_t)512 * 256 * sizeof(bf16), hipMemcpyDeviceToDevice, stream); }

  // =================== FP2 ===================
  k_interp_concat<<<nblk(B * 512, 128), 128, 0, stream>>>(
      l1xyz, l2xyz, l1pts, 320, 320, l2p2, 256, 256, ping, 576, 512, 128, B);
  { bf16* a = run_mlp(ping, (long long)B * 512, 576, fp2L, 2);
    hipMemcpyAsync(l1p2, a, (size_t)B * 512 * 128 * sizeof(bf16), hipMemcpyDeviceToDevice, stream); }

  // =================== FP1 ===================
  k_l0in<<<nblk((long long)B * N * 22, 256), 256, 0, stream>>>(xyz, l0in);
  k_interp_concat<<<nblk(B * N, 128), 128, 0, stream>>>(
      xyz, l1xyz, l0in, 22, 22, l1p2, 128, 128, ping, 160, N, 512, B);
  bf16* l0p = run_mlp(ping, (long long)B * N, 160, fp1L, 2);     // [16384,128]

  // =================== conv1 + conv2 ===================
  bf16* feat = run_mlp(l0p, (long long)B * N, 128, conv1L, 1);   // [16384,64]
  { int cp = 64, kp = 64;
    bf16* wp = (bf16*)alloc((size_t)cp * kp * sizeof(bf16));
    k_packw<<<nblk((long long)cp * kp, 256), 256, 0, stream>>>(c2W, 50, 64, wp, cp, kp);
    launch_gemm(feat, wp, c2b, (float*)d_out, (long long)B * N, kp, cp, 50, 50);
  }
}
